// DeformBottleneck_7069516169324
// MI455X (gfx1250) — matrix operand order repaired and checked by reading, not compiled, not run on hardware
//
#include <hip/hip_runtime.h>

// ---------------- problem dims ----------------
#define BATCH 8
#define C1    256
#define CHN   128
#define C2    256
#define HH    80
#define WW    80
#define HW    (HH*WW)          // 6400
#define NTAP  9
#define MOFF  27               // 3*NTAP
#define KOFF  (CHN*NTAP)       // 1152
#define EPSV  1e-5f

typedef __attribute__((ext_vector_type(16))) __bf16 v16bf;
typedef __attribute__((ext_vector_type(8)))  float  v8f;
typedef __attribute__((ext_vector_type(4)))  unsigned int v4u;
typedef __attribute__((ext_vector_type(8)))  int v8i;
typedef __attribute__((ext_vector_type(4)))  int v4i;

#if __has_builtin(__builtin_amdgcn_tensor_load_to_lds) && __has_builtin(__builtin_amdgcn_s_wait_tensorcnt)
#define HAVE_TDM 1
#else
#define HAVE_TDM 0
#endif

static __device__ __forceinline__ __bf16 f2bf(float f) { return (__bf16)f; }  // v_cvt_pk_bf16_f32
static __device__ __forceinline__ float siluf(float v) { return v / (1.f + __expf(-v)); }

// A fragment: 16x32 bf16 contiguous-k layout: lane holds row m=lane&15;
// K in [kb,kb+16), kb = 16*(lane>>4). Source f32 row-major [M][lda].
static __device__ __forceinline__ v16bf load_a(const float* __restrict__ A, int lda,
                                               int m0, int k0, int lane) {
  int m  = m0 + (lane & 15);
  int kb = (lane >> 4) * 16;
  const float* row = A + (size_t)m * lda + k0;
  v16bf a;
#pragma unroll
  for (int i = 0; i < 8; ++i) { a[i] = f2bf(row[kb + i]); a[8 + i] = f2bf(row[kb + 8 + i]); }
  return a;
}
static __device__ __forceinline__ v16bf load_a_g(const float* __restrict__ A, int lda, int Mv,
                                                 int m0, int k0, int lane) {
  int m  = m0 + (lane & 15);
  int kb = (lane >> 4) * 16;
  v16bf a;
  if (m < Mv) {
    const float* row = A + (size_t)m * lda + k0;
#pragma unroll
    for (int i = 0; i < 8; ++i) { a[i] = f2bf(row[kb + i]); a[8 + i] = f2bf(row[kb + 8 + i]); }
  } else {
#pragma unroll
    for (int i = 0; i < 16; ++i) a[i] = f2bf(0.f);
  }
  return a;
}
// B fragment: 32x16 bf16; lane holds column n=n0+(lane&15), K in [kb,kb+16), kb=16*(lane>>4).
static __device__ __forceinline__ v16bf load_b_f32(const float* __restrict__ B, int ldb,
                                                   int k0, int n0, int lane) {
  int n  = n0 + (lane & 15);
  int kb = (lane >> 4) * 16;
  v16bf b;
#pragma unroll
  for (int i = 0; i < 16; ++i) b[i] = f2bf(B[(size_t)(k0 + kb + i) * ldb + n]);
  return b;
}

// ---------------- kernel 1: cv1 1x1 conv + BN1 + SiLU (wave = 16x64 strip) ----------------
__global__ __launch_bounds__(256) void cv1_kernel(
    const float* __restrict__ x,  const float* __restrict__ w,
    const float* __restrict__ g,  const float* __restrict__ bb,
    const float* __restrict__ mu, const float* __restrict__ var,
    float* __restrict__ y1f, __bf16* __restrict__ y1bf)
{
  const int lane = threadIdx.x & 31;
  const int wid  = blockIdx.x * 8 + (threadIdx.x >> 5);
  const int NB = HW / 64, MT = CHN / 16;        // 100, 8
  int b  = wid / (MT * NB);
  int r  = wid % (MT * NB);
  int m0 = (r / NB) * 16, n0 = (r % NB) * 64;
  const float* Bp = x + (size_t)b * C1 * HW;
  v8f a0 = {}, a1 = {}, a2 = {}, a3 = {};
  for (int k0 = 0; k0 < C1; k0 += 32) {
    __builtin_prefetch(w + (size_t)(m0 + (lane & 15)) * C1 + k0 + 32, 0, 3);
    v16bf af = load_a(w, C1, m0, k0, lane);
    v16bf b0 = load_b_f32(Bp, HW, k0, n0,      lane);
    v16bf b1 = load_b_f32(Bp, HW, k0, n0 + 16, lane);
    v16bf b2 = load_b_f32(Bp, HW, k0, n0 + 32, lane);
    v16bf b3 = load_b_f32(Bp, HW, k0, n0 + 48, lane);
    a0 = __builtin_amdgcn_wmma_f32_16x16x32_bf16(false, af, false, b0, (short)0, a0, false, false);
    a1 = __builtin_amdgcn_wmma_f32_16x16x32_bf16(false, af, false, b1, (short)0, a1, false, false);
    a2 = __builtin_amdgcn_wmma_f32_16x16x32_bf16(false, af, false, b2, (short)0, a2, false, false);
    a3 = __builtin_amdgcn_wmma_f32_16x16x32_bf16(false, af, false, b3, (short)0, a3, false, false);
  }
  int nl = lane & 15, mh = (lane >> 4) * 8;
  v8f accs[4] = {a0, a1, a2, a3};
#pragma unroll
  for (int v = 0; v < 8; ++v) {
    int m = m0 + mh + v;
    float inv = g[m] * rsqrtf(var[m] + EPSV);
    float sh  = bb[m] - mu[m] * inv;
    size_t rowbase = ((size_t)b * CHN + m) * HW + n0 + nl;
#pragma unroll
    for (int t = 0; t < 4; ++t) {
      float val = siluf(accs[t][v] * inv + sh);
      y1f[rowbase + 16 * t]  = val;
      y1bf[rowbase + 16 * t] = f2bf(val);
    }
  }
}

// ---------------- kernel 2: 3x3 offset conv (implicit im2col GEMM) ----------------
__global__ __launch_bounds__(256) void off_kernel(
    const float* __restrict__ ow, const float* __restrict__ ob,
    const __bf16* __restrict__ y1bf, float* __restrict__ raw)
{
  const int lane = threadIdx.x & 31;
  const int wid  = blockIdx.x * 8 + (threadIdx.x >> 5);
  const int NT = HW / 16;
  int b  = wid / (2 * NT);
  int r  = wid % (2 * NT);
  int m0 = (r / NT) * 16, n0 = (r % NT) * 16;
  const __bf16* Yb = y1bf + (size_t)b * CHN * HW;
  int n  = n0 + (lane & 15);
  int py = n / WW, px = n % WW;
  int kb = (lane >> 4) * 16;
  v8f acc = {};
  for (int k0 = 0; k0 < KOFF; k0 += 32) {
    v16bf af = load_a_g(ow, KOFF, MOFF, m0, k0, lane);
    v16bf bf;
#pragma unroll
    for (int i = 0; i < 16; ++i) {
      int k = k0 + kb + i;
      int c = k / NTAP;
      int t = k - c * NTAP;
      int yy = py + (t / 3) - 1;
      int xx = px + (t % 3) - 1;
      float v = 0.f;
      if (yy >= 0 && yy < HH && xx >= 0 && xx < WW)
        v = (float)Yb[(size_t)c * HW + yy * WW + xx];
      bf[i] = f2bf(v);
    }
    acc = __builtin_amdgcn_wmma_f32_16x16x32_bf16(false, af, false, bf, (short)0, acc, false, false);
  }
  int nl = lane & 15, mh = (lane >> 4) * 8;
#pragma unroll
  for (int v = 0; v < 8; ++v) {
    int m = m0 + mh + v;
    if (m >= MOFF) continue;
    float val = acc[v] + ob[m];
    if (m >= 2 * NTAP) val = 1.f / (1.f + __expf(-val));   // mask channels: sigmoid
    raw[((size_t)b * MOFF + m) * HW + n0 + nl] = val;
  }
}

// ---------------- kernel 3: modulated deformable conv + BN2 + SiLU ----------------
// Block: 32 pixels, all 128 out channels. Weight K-tile (128x32 f32) staged in LDS by TDM.
__global__ __launch_bounds__(256) void dcn_kernel(
    const float* __restrict__ dw, const float* __restrict__ db,
    const float* __restrict__ g,  const float* __restrict__ bb,
    const float* __restrict__ mu, const float* __restrict__ var,
    const float* __restrict__ y1f, const float* __restrict__ raw,
    __bf16* __restrict__ y2bf)
{
  __shared__ float  sA[CHN][32];        // 16KB: weight K-tile (TDM destination)
  __shared__ float  swt[NTAP][32][4];   // corner weights * mask * validity
  __shared__ int    sy0[NTAP][32];
  __shared__ int    sx0[NTAP][32];
  __shared__ __bf16 samp[32][32];       // 32(k) x 32(pixels) sample tile
  const int tid  = threadIdx.x;
  const int lane = tid & 31, wv = tid >> 5;
  const int b  = blockIdx.x / (HW / 32);
  const int nt = blockIdx.x % (HW / 32);
  const int n0 = nt * 32;
  const float* rawb = raw + (size_t)b * MOFF * HW;

  for (int i = tid; i < NTAP * 32; i += 256) {
    int t = i >> 5, p = i & 31;
    int n = n0 + p;
    int oy = n / WW, ox = n % WW;
    float dy = rawb[(size_t)(2 * t)     * HW + n];
    float dx = rawb[(size_t)(2 * t + 1) * HW + n];
    float mk = rawb[(size_t)(2 * NTAP + t) * HW + n];
    float pyv = (float)(oy - 1 + t / 3) + dy;
    float pxv = (float)(ox - 1 + t % 3) + dx;
    float y0f = floorf(pyv), x0f = floorf(pxv);
    float fy = pyv - y0f, fx = pxv - x0f;
    int y0 = (int)y0f, x0 = (int)x0f;
    float v00 = (y0   >= 0 && y0   < HH && x0   >= 0 && x0   < WW) ? 1.f : 0.f;
    float v01 = (y0   >= 0 && y0   < HH && x0+1 >= 0 && x0+1 < WW) ? 1.f : 0.f;
    float v10 = (y0+1 >= 0 && y0+1 < HH && x0   >= 0 && x0   < WW) ? 1.f : 0.f;
    float v11 = (y0+1 >= 0 && y0+1 < HH && x0+1 >= 0 && x0+1 < WW) ? 1.f : 0.f;
    swt[t][p][0] = (1.f - fy) * (1.f - fx) * mk * v00;
    swt[t][p][1] = (1.f - fy) * fx         * mk * v01;
    swt[t][p][2] = fy         * (1.f - fx) * mk * v10;
    swt[t][p][3] = fy         * fx         * mk * v11;
    sy0[t][p] = y0; sx0[t][p] = x0;
  }
  __syncthreads();

  const float* Yb = y1f + (size_t)b * CHN * HW;
  const int m0 = wv * 16;
  v8f acc0 = {}, acc1 = {};
  for (int kt = 0; kt < KOFF / 32; ++kt) {       // 36 steps
#if HAVE_TDM
    if (wv == 0) {
      // Tensor DMA: 32(k) x 128(rows) f32 tile of dcn_w -> sA, row stride KOFF
      unsigned lds = (unsigned)(unsigned long long)(void*)&sA[0][0];
      unsigned long long ga = (unsigned long long)(const void*)(dw + (size_t)kt * 32);
      v4u g0;
      g0[0] = 1u;                                          // count=1, user mode
      g0[1] = lds;                                         // lds_addr
      g0[2] = (unsigned)ga;                                // global_addr[31:0]
      g0[3] = (unsigned)((ga >> 32) & 0x1FFFFFFu) | (2u << 30); // addr[56:32] | type=2
      v8i g1;
      g1[0] = (2 << 16);                                   // data_size=4B, no mask
      g1[1] = (int)((KOFF & 0xFFFF) << 16);                // tensor_dim0[15:0]
      g1[2] = (int)((KOFF >> 16) | ((CHN & 0xFFFF) << 16));// dim0[31:16] | dim1[15:0]
      g1[3] = (int)((CHN >> 16) | (32 << 16));             // dim1[31:16] | tile_dim0=32
      g1[4] = (int)CHN;                                    // tile_dim1=128, tile_dim2=0
      g1[5] = (int)KOFF;                                   // tensor_dim0_stride[31:0]
      g1[6] = 0;                                           // stride hi | dim1_stride lo
      g1[7] = 0;
      v4i z4 = {};
      v8i z8 = {};
      __builtin_amdgcn_tensor_load_to_lds(g0, g1, z4, z4, z8, 0);  // 6-arg toolchain form
    }
#else
    for (int i = tid; i < CHN * 32; i += 256) {
      int rr = i >> 5, cc = i & 31;
      sA[rr][cc] = dw[(size_t)rr * KOFF + kt * 32 + cc];
    }
#endif
    // cooperatively build 32x32 bilinear-sampled bf16 tile (overlaps with TDM)
    for (int e = tid; e < 1024; e += 256) {
      int kk = e >> 5, p = e & 31;
      int k = kt * 32 + kk;
      int c = k / NTAP, t = k - c * NTAP;
      const float* ch = Yb + (size_t)c * HW;
      int y0 = sy0[t][p], x0 = sx0[t][p];
      int ya = min(max(y0,     0), HH - 1), yb2 = min(max(y0 + 1, 0), HH - 1);
      int xa = min(max(x0,     0), WW - 1), xb2 = min(max(x0 + 1, 0), WW - 1);
      float v = swt[t][p][0] * ch[ya * WW + xa]  + swt[t][p][1] * ch[ya * WW + xb2]
              + swt[t][p][2] * ch[yb2 * WW + xa] + swt[t][p][3] * ch[yb2 * WW + xb2];
      samp[kk][p] = f2bf(v);
    }
#if HAVE_TDM
    if (wv == 0) __builtin_amdgcn_s_wait_tensorcnt(0);
#endif
    __syncthreads();
    // A fragment from LDS weight tile
    {
      int mrow = m0 + (lane & 15);
      int ka = (lane >> 4) * 16;
      v16bf af;
#pragma unroll
      for (int i = 0; i < 8; ++i) {
        af[i]     = f2bf(sA[mrow][ka + i]);
        af[8 + i] = f2bf(sA[mrow][ka + 8 + i]);
      }
      int nl = lane & 15, kb = (lane >> 4) * 16;
      v16bf b0, b1;
#pragma unroll
      for (int i = 0; i < 16; ++i) { b0[i] = samp[kb + i][nl]; b1[i] = samp[kb + i][16 + nl]; }
      acc0 = __builtin_amdgcn_wmma_f32_16x16x32_bf16(false, af, false, b0, (short)0, acc0, false, false);
      acc1 = __builtin_amdgcn_wmma_f32_16x16x32_bf16(false, af, false, b1, (short)0, acc1, false, false);
    }
    __syncthreads();
  }
  int nl = lane & 15, mh = (lane >> 4) * 8;
#pragma unroll
  for (int v = 0; v < 8; ++v) {
    int m = m0 + mh + v;
    float inv = g[m] * rsqrtf(var[m] + EPSV);
    float sh  = bb[m] - mu[m] * inv + db[m] * inv;   // fold conv bias through BN scale
    float v0 = siluf(acc0[v] * inv + sh);
    float v1 = siluf(acc1[v] * inv + sh);
    size_t base = ((size_t)b * CHN + m) * HW + n0;
    y2bf[base + nl]      = f2bf(v0);
    y2bf[base + 16 + nl] = f2bf(v1);
  }
}

// ---------------- kernel 4: cv2 1x1 conv + BN3 + SiLU + residual (wave = 16x64) ----------------
__global__ __launch_bounds__(256) void cv2_kernel(
    const float* __restrict__ x,  const float* __restrict__ w,
    const float* __restrict__ g,  const float* __restrict__ bb,
    const float* __restrict__ mu, const float* __restrict__ var,
    const __bf16* __restrict__ y2bf, float* __restrict__ out)
{
  const int lane = threadIdx.x & 31;
  const int wid  = blockIdx.x * 8 + (threadIdx.x >> 5);
  const int NB = HW / 64, MT = C2 / 16;        // 100, 16
  int b  = wid / (MT * NB);
  int r  = wid % (MT * NB);
  int m0 = (r / NB) * 16, n0 = (r % NB) * 64;
  const __bf16* Bp = y2bf + (size_t)b * CHN * HW;
  int nl = lane & 15, kb = (lane >> 4) * 16;
  v8f a0 = {}, a1 = {}, a2 = {}, a3 = {};
  for (int k0 = 0; k0 < CHN; k0 += 32) {
    v16bf af = load_a(w, CHN, m0, k0, lane);
    v16bf b0, b1, b2, b3;
#pragma unroll
    for (int i = 0; i < 16; ++i) {
      size_t rowo = (size_t)(k0 + kb + i) * HW + n0 + nl;
      b0[i] = Bp[rowo];
      b1[i] = Bp[rowo + 16];
      b2[i] = Bp[rowo + 32];
      b3[i] = Bp[rowo + 48];
    }
    a0 = __builtin_amdgcn_wmma_f32_16x16x32_bf16(false, af, false, b0, (short)0, a0, false, false);
    a1 = __builtin_amdgcn_wmma_f32_16x16x32_bf16(false, af, false, b1, (short)0, a1, false, false);
    a2 = __builtin_amdgcn_wmma_f32_16x16x32_bf16(false, af, false, b2, (short)0, a2, false, false);
    a3 = __builtin_amdgcn_wmma_f32_16x16x32_bf16(false, af, false, b3, (short)0, a3, false, false);
  }
  int mh = (lane >> 4) * 8;
  v8f accs[4] = {a0, a1, a2, a3};
#pragma unroll
  for (int v = 0; v < 8; ++v) {
    int m = m0 + mh + v;
    float inv = g[m] * rsqrtf(var[m] + EPSV);
    float sh  = bb[m] - mu[m] * inv;
    size_t rowbase = ((size_t)b * C2 + m) * HW + n0 + nl;
#pragma unroll
    for (int t = 0; t < 4; ++t) {
      float val = siluf(accs[t][v] * inv + sh);
      size_t o = rowbase + 16 * t;
      out[o] = x[o] + val;
    }
  }
}

// ---------------- host launch ----------------
extern "C" void kernel_launch(void* const* d_in, const int* in_sizes, int n_in,
                              void* d_out, int out_size, void* d_ws, size_t ws_size,
                              hipStream_t stream)
{
  const float* x    = (const float*)d_in[0];
  const float* cv1w = (const float*)d_in[1];
  const float* bn1g = (const float*)d_in[2];
  const float* bn1b = (const float*)d_in[3];
  const float* bn1m = (const float*)d_in[4];
  const float* bn1v = (const float*)d_in[5];
  const float* offw = (const float*)d_in[6];
  const float* offb = (const float*)d_in[7];
  const float* dcnw = (const float*)d_in[8];
  const float* dcnb = (const float*)d_in[9];
  const float* bn2g = (const float*)d_in[10];
  const float* bn2b = (const float*)d_in[11];
  const float* bn2m = (const float*)d_in[12];
  const float* bn2v = (const float*)d_in[13];
  const float* cv2w = (const float*)d_in[14];
  const float* bn3g = (const float*)d_in[15];
  const float* bn3b = (const float*)d_in[16];
  const float* bn3m = (const float*)d_in[17];
  const float* bn3v = (const float*)d_in[18];
  float* out = (float*)d_out;
  (void)in_sizes; (void)n_in; (void)out_size; (void)ws_size;

  // workspace layout (~58 MB)
  char* ws = (char*)d_ws;
  size_t o = 0;
  float*  y1f  = (float*)(ws + o);  o += (size_t)BATCH * CHN  * HW * 4;
  __bf16* y1bf = (__bf16*)(ws + o); o += (size_t)BATCH * CHN  * HW * 2;
  float*  raw  = (float*)(ws + o);  o += (size_t)BATCH * MOFF * HW * 4;
  __bf16* y2bf = (__bf16*)(ws + o); o += (size_t)BATCH * CHN  * HW * 2;

  dim3 blk(256);
  cv1_kernel<<<BATCH * (CHN/16) * (HW/64) / 8, blk, 0, stream>>>(
      x, cv1w, bn1g, bn1b, bn1m, bn1v, y1f, y1bf);
  off_kernel<<<BATCH * 2 * (HW/16) / 8, blk, 0, stream>>>(
      offw, offb, y1bf, raw);
  dcn_kernel<<<BATCH * (HW/32), blk, 0, stream>>>(
      dcnw, dcnb, bn2g, bn2b, bn2m, bn2v, y1f, raw, y2bf);
  cv2_kernel<<<BATCH * (C2/16) * (HW/64) / 8, blk, 0, stream>>>(
      x, cv2w, bn3g, bn3b, bn3m, bn3v, y2bf, out);
}
